// Attention_34763465294050
// MI455X (gfx1250) — matrix operand order repaired and checked by reading, not compiled, not run on hardware
//
#include <hip/hip_runtime.h>

typedef __bf16 bf16;
typedef __attribute__((ext_vector_type(16))) __bf16 v16bf;
typedef __attribute__((ext_vector_type(8)))  __bf16 v8bf;
typedef __attribute__((ext_vector_type(2)))  __bf16 v2bf;
typedef __attribute__((ext_vector_type(8)))  float  v8f;
typedef __attribute__((ext_vector_type(2)))  float  v2f;
typedef __attribute__((ext_vector_type(8)))  unsigned u32x8;

#define NUM_HEADS 8
#define HEAD_DIM  64
#define EMBED     512
#define BATCH     4
#define SEQ       2048
#define NROWS     (BATCH * SEQ)   // 8192

// workspace layout (bytes)
#define KP_OFF   (0)
#define VPT_OFF  (8u << 20)    // Kp  [B,H,S,64] bf16 (8 MB)
#define AO_OFF   (16u << 20)   // VpT [B,H,64,S] bf16 (8 MB)
#define WO_OFF   (24u << 20)   // AO  [B*S,512]  bf16 (8 MB); then 512KB Wo bf16

#define LOG2E 1.4426950408889634f

__device__ __forceinline__ v8bf cvt8(v8f x) { return __builtin_convertvector(x, v8bf); }

__device__ __forceinline__ v16bf cat16(v8bf lo, v8bf hi) {
  return __builtin_shufflevector(lo, hi, 0, 1, 2, 3, 4, 5, 6, 7,
                                         8, 9, 10, 11, 12, 13, 14, 15);
}

__device__ __forceinline__ unsigned pack2(float a, float b) {
  v2f f; f[0] = a; f[1] = b;
  return __builtin_bit_cast(unsigned, __builtin_convertvector(f, v2bf));
}

// ---- WMMA operand loaders (row/col per lane as in the ISA; k contiguous, k = 16*(L>>4) + s, same order in A and B) ----

// A operand 16x32 bf16 row-major source: lane L = row (L&15);
// 16 contiguous K at kk+(L>>4)*16
__device__ __forceinline__ v16bf load_a_bf16(const bf16* base, int stride, int lane, int kk) {
  const bf16* p = base + (size_t)(lane & 15) * stride + kk + ((lane >> 4) << 4);
  return cat16(*(const v8bf*)p, *(const v8bf*)(p + 8));
}

__device__ __forceinline__ v16bf load_a_f32(const float* base, int stride, int lane, int kk) {
  const float* p = base + (size_t)(lane & 15) * stride + kk + ((lane >> 4) << 4);
  return cat16(cvt8(*(const v8f*)p), cvt8(*(const v8f*)(p + 8)));
}

// B operand 32x16 bf16: lane L = column (L&15); 16 contiguous K at kk+(L>>4)*16,
// from "B-transposed row-major" memory (row n holds K contiguous).
__device__ __forceinline__ v16bf load_b_bf16(const bf16* base, int stride, int lane, int kk) {
  const bf16* p = base + (size_t)(lane & 15) * stride + kk + ((lane >> 4) << 4);
  return *(const v16bf*)p;
}

__device__ __forceinline__ v16bf load_b_f32(const float* base, int stride, int lane, int kk) {
  const float* p = base + (size_t)(lane & 15) * stride + kk + ((lane >> 4) << 4);
  return cat16(cvt8(*(const v8f*)p), cvt8(*(const v8f*)(p + 8)));
}

__device__ __forceinline__ v8f wmma_bf(v16bf a, v16bf b, v8f c) {
  return __builtin_amdgcn_wmma_f32_16x16x32_bf16(false, a, false, b, (short)0, c, false, false);
}

// ---------------- kernel 1: K' (computed transposed) + V' (transposed store) ----------------
__global__ __launch_bounds__(256) void proj_kv_kernel(const float* __restrict__ X,
    const float* __restrict__ Wk, const float* __restrict__ Wv,
    bf16* __restrict__ Kp, bf16* __restrict__ VpT) {
  int lane = threadIdx.x & 31;
  int hL = lane >> 4;
  int wave = threadIdx.x >> 5;
  int tile = blockIdx.x * 8 + wave;          // 0..511 -> 16 rows each
  int h = blockIdx.y;
  int r0 = tile * 16;                        // global row 0..8191
  int b = r0 >> 11;
  int s0 = r0 & (SEQ - 1);
  const float* Xr = X + (size_t)r0 * EMBED + h * HEAD_DIM;

  v16bf ax0 = load_a_f32(Xr, EMBED, lane, 0);   // X as A (rows = s)
  v16bf ax1 = load_a_f32(Xr, EMBED, lane, 32);
  v16bf bx0 = load_b_f32(Xr, EMBED, lane, 0);   // X^T as B (cols = s)
  v16bf bx1 = load_b_f32(Xr, EMBED, lane, 32);

  size_t hb = (size_t)b * NUM_HEADS + h;
  bf16* kout = Kp + (hb * SEQ + s0) * HEAD_DIM;     // [s][64] row-major
  bf16* vout = VpT + hb * HEAD_DIM * SEQ + s0;      // [e][S], col offset s0

#pragma unroll
  for (int n = 0; n < 4; ++n) {
    // K'^T tile = Wk x X^T : C(M=e, N=s) -> contiguous-e store at row s
    v8f ck = {};
    ck = wmma_bf(load_a_f32(Wk + n * 16 * HEAD_DIM, HEAD_DIM, lane, 0),  bx0, ck);
    ck = wmma_bf(load_a_f32(Wk + n * 16 * HEAD_DIM, HEAD_DIM, lane, 32), bx1, ck);
    *(v8bf*)(kout + (size_t)(lane & 15) * HEAD_DIM + n * 16 + 8 * hL) = cvt8(ck);
    // V' tile = X x Wv^T : C(M=s, N=e) -> contiguous-s store at row e
    v8f cv = {};
    cv = wmma_bf(ax0, load_b_f32(Wv + n * 16 * HEAD_DIM, HEAD_DIM, lane, 0),  cv);
    cv = wmma_bf(ax1, load_b_f32(Wv + n * 16 * HEAD_DIM, HEAD_DIM, lane, 32), cv);
    *(v8bf*)(vout + (size_t)(n * 16 + (lane & 15)) * SEQ + 8 * hL) = cvt8(cv);
  }
}

// ---------------- kernel 2: Wo f32 -> bf16 (vectorized) ----------------
__global__ __launch_bounds__(256) void cvt_wo_kernel(const float* __restrict__ Wo,
                                                     bf16* __restrict__ Wo16) {
  int i = (blockIdx.x * 256 + threadIdx.x) * 8;
  *(v8bf*)(Wo16 + i) = cvt8(*(const v8f*)(Wo + i));
}

// ---------------- kernel 3: flash attention, transposed math, LDS-staged K/V ----------------
__global__ __launch_bounds__(256) void flash_kernel(const float* __restrict__ X,
    const float* __restrict__ Wq, const bf16* __restrict__ Kp,
    const bf16* __restrict__ VpT, bf16* __restrict__ AO) {
  __shared__ bf16 Qs[8][16][64];              // per-wave Q' [q][e] (setup only)   16KB
  __shared__ bf16 Ks[2][64][64];              // K tile double buffer [key][e]     16KB
  __shared__ bf16 Vs[2][64][64];              // V tile double buffer [e][key]     16KB
  int lane = threadIdx.x & 31;
  int hL = lane >> 4;
  bool hi = hL != 0;
  int wave = threadIdx.x >> 5;
  int h = blockIdx.y, b = blockIdx.z;
  int qrow0 = blockIdx.x * 128 + wave * 16;

  const float* Xr = X + ((size_t)b * SEQ + qrow0) * EMBED + h * HEAD_DIM;
  bf16 (*Qb)[64] = Qs[wave];

  size_t hb = (size_t)b * NUM_HEADS + h;
  const bf16* Kb = Kp + hb * SEQ * HEAD_DIM;
  const bf16* Vb = VpT + hb * HEAD_DIM * SEQ;

  // cooperative staging coordinates: 256 threads x 4 x 16B = 16KB per (K,V) pair
  int srow = threadIdx.x >> 3;                // 0..31
  int sch  = (threadIdx.x & 7) * 8;           // 0..56

  // stage tile 0 directly
  *(v8bf*)&Ks[0][srow][sch]      = *(const v8bf*)(Kb + (size_t)srow * HEAD_DIM + sch);
  *(v8bf*)&Ks[0][srow + 32][sch] = *(const v8bf*)(Kb + (size_t)(srow + 32) * HEAD_DIM + sch);
  *(v8bf*)&Vs[0][srow][sch]      = *(const v8bf*)(Vb + (size_t)srow * SEQ + sch);
  *(v8bf*)&Vs[0][srow + 32][sch] = *(const v8bf*)(Vb + (size_t)(srow + 32) * SEQ + sch);

  // Q'^T = Wq x X^T, scale (1/8)*log2(e) folded; store Q' row-major to LDS (b128 stores)
  const float qscale = 0.125f * LOG2E;
  v16bf bx0 = load_b_f32(Xr, EMBED, lane, 0);
  v16bf bx1 = load_b_f32(Xr, EMBED, lane, 32);
#pragma unroll
  for (int n = 0; n < 4; ++n) {
    v8f c = {};
    c = wmma_bf(load_a_f32(Wq + n * 16 * HEAD_DIM, HEAD_DIM, lane, 0),  bx0, c);
    c = wmma_bf(load_a_f32(Wq + n * 16 * HEAD_DIM, HEAD_DIM, lane, 32), bx1, c);
    *(v8bf*)(&Qb[lane & 15][n * 16 + 8 * hL]) = cvt8(c * qscale);
  }
  v16bf qb0 = load_b_bf16(&Qb[0][0], 64, lane, 0);   // Q'^T as B operand (col = query)
  v16bf qb1 = load_b_bf16(&Qb[0][0], 64, lane, 32);

  __syncthreads();

  float m = -1e30f, l = 0.f;                  // per-lane: stats of query (lane&15)
  v8f O[4] = {};                              // O^T tiles: d = n*16 + r + 8*hL, q = lane&15

  for (int kt = 0; kt < SEQ; kt += 64) {
    int p = (kt >> 6) & 1;
    // issue next tile's global loads first; commit to LDS after compute
    v8bf k0, k1, v0, v1;
    bool have_next = (kt + 64 < SEQ);
    if (have_next) {
      const bf16* kn = Kb + (size_t)(kt + 64) * HEAD_DIM;
      const bf16* vn = Vb + (kt + 64);
      k0 = *(const v8bf*)(kn + (size_t)srow * HEAD_DIM + sch);
      k1 = *(const v8bf*)(kn + (size_t)(srow + 32) * HEAD_DIM + sch);
      v0 = *(const v8bf*)(vn + (size_t)srow * SEQ + sch);
      v1 = *(const v8bf*)(vn + (size_t)(srow + 32) * SEQ + sch);
    }

    // S^T tiles: C(M = key, N = query), operands from LDS
    v8f St[4];
#pragma unroll
    for (int j = 0; j < 4; ++j) {
      const bf16* kb = &Ks[p][j * 16][0];
      v8f c = {};
      c = wmma_bf(load_a_bf16(kb, 64, lane, 0),  qb0, c);
      c = wmma_bf(load_a_bf16(kb, 64, lane, 32), qb1, c);
      St[j] = c;
    }
    // softmax stats: in-lane reduce over 8 comps x 4 tiles, one half-swap shuffle
    float mloc = -1e30f;
#pragma unroll
    for (int j = 0; j < 4; ++j)
#pragma unroll
      for (int r = 0; r < 8; ++r) mloc = fmaxf(mloc, St[j][r]);
    mloc = fmaxf(mloc, __shfl_xor(mloc, 16, 32));
    float mx = fmaxf(m, mloc);
    float alpha = exp2f(m - mx);
    // P = exp2(S - mx), packed to bf16 pairs (keys along packing dim)
    unsigned pk[4][4];
    float rsum = 0.f;
#pragma unroll
    for (int j = 0; j < 4; ++j) {
      float pv[8];
#pragma unroll
      for (int r = 0; r < 8; ++r) { pv[r] = exp2f(St[j][r] - mx); rsum += pv[r]; }
#pragma unroll
      for (int i = 0; i < 4; ++i) pk[j][i] = pack2(pv[2 * i], pv[2 * i + 1]);
    }
    rsum += __shfl_xor(rsum, 16, 32);
    l = l * alpha + rsum;
    m = mx;
#pragma unroll
    for (int n = 0; n < 4; ++n) O[n] *= alpha;
    // O^T += V'^T x P : assemble P as B operand via half-swap shuffles, no LDS
#pragma unroll
    for (int c2 = 0; c2 < 2; ++c2) {
      int j0 = c2 * 2;
      u32x8 dv;
#pragma unroll
      for (int i = 0; i < 4; ++i) {
        unsigned x = __shfl_xor(pk[j0][i], 16, 32);       // partner's tile j0
        unsigned y = __shfl_xor(pk[j0 + 1][i], 16, 32);   // partner's tile j0+1
        dv[i]     = hi ? y : pk[j0][i];
        dv[i + 4] = hi ? pk[j0 + 1][i] : x;
      }
      v16bf pb = __builtin_bit_cast(v16bf, dv);
#pragma unroll
      for (int n = 0; n < 4; ++n)
        O[n] = wmma_bf(load_a_bf16(&Vs[p][n * 16][0], 64, lane, c2 * 32), pb, O[n]);
    }

    // commit staged tile and synchronize (also guards next iteration's overwrite)
    if (have_next) {
      int q = p ^ 1;
      *(v8bf*)&Ks[q][srow][sch]      = k0;
      *(v8bf*)&Ks[q][srow + 32][sch] = k1;
      *(v8bf*)&Vs[q][srow][sch]      = v0;
      *(v8bf*)&Vs[q][srow + 32][sch] = v1;
    }
    __syncthreads();
  }

  // epilogue: per-lane 1/l scale, contiguous packed stores (one v8bf per tile)
  float inv = 1.0f / l;
  bf16* out = AO + ((size_t)b * SEQ + qrow0 + (lane & 15)) * EMBED + h * HEAD_DIM;
#pragma unroll
  for (int n = 0; n < 4; ++n)
    *(v8bf*)(out + n * 16 + 8 * hL) = cvt8(O[n] * inv);
}

// ---------------- kernel 4: output projection, transposed: out^T = Wo x AO^T ----------------
__global__ __launch_bounds__(256) void oproj_kernel(const bf16* __restrict__ AO,
    const bf16* __restrict__ Wo16, const float* __restrict__ bo,
    float* __restrict__ out) {
  int lane = threadIdx.x & 31;
  int hL = lane >> 4;
  int wave = threadIdx.x >> 5;
  int row0 = blockIdx.y * 128 + wave * 16;
  int col0 = blockIdx.x * 64;
  v8f c[4] = {};
  for (int kk = 0; kk < EMBED; kk += 32) {
    v16bf bA = load_b_bf16(AO + (size_t)row0 * EMBED, EMBED, lane, kk);  // AO^T (col = out row)
#pragma unroll
    for (int n = 0; n < 4; ++n)
      c[n] = wmma_bf(load_a_bf16(Wo16 + (size_t)(col0 + n * 16) * EMBED, EMBED, lane, kk),
                     bA, c[n]);
  }
  // C(M = out col, N = out row): contiguous-col stores + vector bias add
  float* orow = out + (size_t)(row0 + (lane & 15)) * EMBED + col0;
#pragma unroll
  for (int n = 0; n < 4; ++n) {
    v8f bb = *(const v8f*)(bo + col0 + n * 16 + 8 * hL);
    *(v8f*)(orow + n * 16 + 8 * hL) = c[n] + bb;
  }
}

extern "C" void kernel_launch(void* const* d_in, const int* in_sizes, int n_in,
                              void* d_out, int out_size, void* d_ws, size_t ws_size,
                              hipStream_t stream) {
  (void)in_sizes; (void)n_in; (void)out_size; (void)ws_size;
  const float* q  = (const float*)d_in[0];
  // d_in[1]=keys, d_in[2]=values are unused (reference derives k/v from queries)
  const float* Wq = (const float*)d_in[3];
  const float* Wk = (const float*)d_in[4];
  const float* Wv = (const float*)d_in[5];
  const float* Wo = (const float*)d_in[6];
  const float* bo = (const float*)d_in[7];

  char* ws = (char*)d_ws;
  bf16* Kp   = (bf16*)(ws + KP_OFF);
  bf16* VpT  = (bf16*)(ws + VPT_OFF);
  bf16* AO   = (bf16*)(ws + AO_OFF);
  bf16* Wo16 = (bf16*)(ws + WO_OFF);

  proj_kv_kernel<<<dim3(64, NUM_HEADS), 256, 0, stream>>>(q, Wk, Wv, Kp, VpT);
  cvt_wo_kernel<<<dim3(EMBED * EMBED / (256 * 8)), 256, 0, stream>>>(Wo, Wo16);
  flash_kernel<<<dim3(SEQ / 128, NUM_HEADS, BATCH), 256, 0, stream>>>(q, Wq, Kp, VpT, AO);
  oproj_kernel<<<dim3(EMBED / 64, NROWS / 128), 256, 0, stream>>>(AO, Wo16, bo, (float*)d_out);
}
